// GNNRoIFusion_44418551775895
// MI455X (gfx1250) — matrix operand order repaired and checked by reading, not compiled, not run on hardware
//
#include <hip/hip_runtime.h>

typedef __bf16 bf16_t;
typedef __attribute__((ext_vector_type(16))) __bf16 v16bf;
typedef __attribute__((ext_vector_type(8)))  float  v8f;
typedef __attribute__((ext_vector_type(4)))  unsigned int u32x4;
typedef __attribute__((ext_vector_type(4)))  int i32x4;
typedef __attribute__((ext_vector_type(8)))  int i32x8;

#define CDIM   256
#define BATCH  2
#define HH     128
#define WW     128
#define HWSZ   (HH * WW)            // 16384
#define NPIX   (BATCH * HWSZ)       // 32768
#define NNODE  4
#define NROWS  (NPIX * NNODE)       // 131072
#define PADW   (WW + 2)             // 130
#define PADH   (HH + 2)

// ---------------- workspace layout (bytes) ----------------
#define XB_OFF    ((size_t)0)                         // bf16 [131072,256]   64 MiB
#define XL_OFF    ((size_t)67108864)                  // bf16 [131072,256]   64 MiB
#define XR_OFF    ((size_t)134217728)                 // bf16 [131072,256]   64 MiB
#define MEAN_OFF  ((size_t)201326592)                 // bf16 [32768,256]    16 MiB
#define H1_OFF    ((size_t)218103808)                 // bf16 [32768,256]    16 MiB
#define PAD_OFF   ((size_t)234881024)                 // bf16 [2,130,130,256]
#define WT_OFF    ((size_t)252186624)                 // bf16 15 x [256,256]

union AV { uint4 u[2]; v16bf v; };
union U8 { uint4 u; bf16_t h[8]; };

// ---------------- TDM availability ----------------
#if defined(__has_builtin)
#if __has_builtin(__builtin_amdgcn_tensor_load_to_lds)
#define HAVE_TDM 1
#endif
#endif
#ifndef HAVE_TDM
#define HAVE_TDM 0
#endif

#if HAVE_TDM
// 2D TDM load: tile [tile1 rows x tile0 elems] of bf16 from a row-major tensor
// (row length dim0, stride stride0, dim1 rows) into contiguous LDS at lds_off.
// D# bit layout per CDNA5 ISA ch.8 (group0: count/lds/global/type, group1: dims).
__device__ __forceinline__ void tdm_load_2d(const void* gaddr, unsigned lds_off,
                                            unsigned dim0, unsigned dim1,
                                            unsigned tile0, unsigned tile1,
                                            unsigned stride0)
{
    unsigned long long ga = (unsigned long long)gaddr;
    u32x4 g0;
    g0[0] = 1u;                                          // count=1, user mode
    g0[1] = lds_off;                                     // lds_addr (bytes)
    g0[2] = (unsigned)(ga & 0xFFFFFFFFu);                // global_addr[31:0]
    g0[3] = (unsigned)((ga >> 32) & 0x01FFFFFFu)         // global_addr[56:32]
          | (2u << 30);                                  // type=2 ("image")
    i32x8 g1;
    g1[0] = (int)(1u << 16);                             // data_size=1 (2B), mask=0
    g1[1] = (int)((dim0 & 0xFFFFu) << 16);               // tensor_dim0[15:0]
    g1[2] = (int)((dim0 >> 16) | ((dim1 & 0xFFFFu) << 16)); // dim0[31:16], dim1[15:0]
    g1[3] = (int)((dim1 >> 16) | (tile0 << 16));         // dim1[31:16], tile_dim0
    g1[4] = (int)(tile1 & 0xFFFFu);                      // tile_dim1 (tile_dim2=0)
    g1[5] = (int)stride0;                                // tensor_dim0_stride[31:0]
    g1[6] = 0;                                           // stride0[47:32], stride1[15:0]
    g1[7] = 0;
    i32x4 gz = {0, 0, 0, 0};
#if __clang_major__ >= 23
    i32x8 gz8 = {0, 0, 0, 0, 0, 0, 0, 0};
    __builtin_amdgcn_tensor_load_to_lds(g0, g1, gz, gz, gz8, 0);
#else
    __builtin_amdgcn_tensor_load_to_lds(g0, g1, gz, gz, 0);
#endif
}
#endif

// ---------------- weight conversion ----------------
__global__ __launch_bounds__(256) void wcvt_dense_kernel(
    const float* __restrict__ w0, const float* __restrict__ w1,
    const float* __restrict__ w2, const float* __restrict__ w3,
    const float* __restrict__ w4, const float* __restrict__ w5,
    bf16_t* __restrict__ out)
{
    int which = blockIdx.y;
    const float* w = (which == 0) ? w0 : (which == 1) ? w1 : (which == 2) ? w2
                   : (which == 3) ? w3 : (which == 4) ? w4 : w5;
    int idx = blockIdx.x * 256 + threadIdx.x;
    int n = idx >> 8, k = idx & 255;
    out[(size_t)which * 65536 + idx] = (bf16_t)w[k * 256 + n];
}

__global__ __launch_bounds__(256) void wcvt_conv_kernel(
    const float* __restrict__ cw, bf16_t* __restrict__ out)
{
    int t = blockIdx.y;
    int idx = blockIdx.x * 256 + threadIdx.x;
    out[(size_t)t * 65536 + idx] = (bf16_t)cw[(size_t)idx * 9 + t];
}

// ---------------- gather modal nodes + mean ----------------
__global__ __launch_bounds__(256) void build_nodes_kernel(
    const float* __restrict__ m0, const float* __restrict__ m1,
    const float* __restrict__ m2, bf16_t* __restrict__ Xb,
    bf16_t* __restrict__ Mean)
{
    int idx = blockIdx.x * 256 + threadIdx.x;
    int c = idx / NPIX;
    int p = idx - c * NPIX;
    int b = p >> 14;
    int rem = p & (HWSZ - 1);
    size_t src = (size_t)b * CDIM * HWSZ + (size_t)c * HWSZ + rem;
    float v0 = m0[src], v1 = m1[src], v2 = m2[src];
    size_t row = (size_t)p * NNODE * CDIM;
    Xb[row + 1 * CDIM + c] = (bf16_t)v0;
    Xb[row + 2 * CDIM + c] = (bf16_t)v1;
    Xb[row + 3 * CDIM + c] = (bf16_t)v2;
    Mean[(size_t)p * CDIM + c] = (bf16_t)((v0 + v1 + v2) * (1.0f / 3.0f));
}

// ---------------- WMMA GEMM: out = act(A @ Wt^T + bias) ----------------
// Block: 256 thr = 8 waves; tile 64 rows x 256 cols; wave 16x128.
// A & B double-buffered in LDS; TDM stage s+1 issued while stage s computes
// (s_wait_tensorcnt 2 = only the just-issued pair still in flight).
__global__ __launch_bounds__(256) void gemm_bf16_kernel(
    const bf16_t* __restrict__ A, const bf16_t* __restrict__ Wt,
    const float* __restrict__ bias, bf16_t* __restrict__ Out,
    int outRowScale, int doRelu)
{
    __shared__ __align__(16) bf16_t As[2][64][32];     //  8 KiB
    __shared__ __align__(16) bf16_t Bs[2][256][32];    // 32 KiB
    int tid = threadIdx.x;
    int lane = tid & 31;
    int mSub = (tid >> 5) & 3, nHalf = tid >> 7;
    int laneRow = lane & 15, laneHi = lane >> 4;
    int rowBase = blockIdx.x * 64;

    v8f acc[8];
    for (int s = 0; s < 8; ++s) acc[s] = v8f{0.f,0.f,0.f,0.f,0.f,0.f,0.f,0.f};

#if HAVE_TDM
    if (tid == 0) {
        tdm_load_2d(A + (size_t)rowBase * 256,
                    (unsigned)(unsigned long long)&As[0][0][0],
                    256, 0x40000000u, 32, 64, 256);
        tdm_load_2d(Wt,
                    (unsigned)(unsigned long long)&Bs[0][0][0],
                    256, 256, 32, 256, 256);
    }
#endif
    for (int step = 0; step < 8; ++step) {
        int cur = step & 1;
#if HAVE_TDM
        if (tid == 0) {
            if (step + 1 < 8) {
                int kn = (step + 1) * 32;
                tdm_load_2d(A + (size_t)rowBase * 256 + kn,
                            (unsigned)(unsigned long long)&As[1 - cur][0][0],
                            256, 0x40000000u, 32, 64, 256);
                tdm_load_2d(Wt + kn,
                            (unsigned)(unsigned long long)&Bs[1 - cur][0][0],
                            256, 256, 32, 256, 256);
                __builtin_amdgcn_s_wait_tensorcnt(2);   // stage `step` done
            } else {
                __builtin_amdgcn_s_wait_tensorcnt(0);
            }
        }
#else
        {
            int k0 = step * 32;
            int r = tid >> 2, cpos = (tid & 3) * 8;
            const bf16_t* asrc = A + (size_t)(rowBase + r) * 256 + k0 + cpos;
            *(uint4*)&As[cur][r][cpos] = *(const uint4*)asrc;
            if (k0 + 32 < 256) __builtin_prefetch((const char*)(asrc + 32), 0, 1);
            const bf16_t* bsrc = Wt + (size_t)tid * 256 + k0;
            *(uint4*)&Bs[cur][tid][0]  = *(const uint4*)(bsrc);
            *(uint4*)&Bs[cur][tid][8]  = *(const uint4*)(bsrc + 8);
            *(uint4*)&Bs[cur][tid][16] = *(const uint4*)(bsrc + 16);
            *(uint4*)&Bs[cur][tid][24] = *(const uint4*)(bsrc + 24);
        }
#endif
        __syncthreads();

        AV af;   // ISA 16-bit A 16x32 layout: row=lane%16, K chunks by lane/16
        {
            int r = mSub * 16 + laneRow;
            af.u[0] = *(const uint4*)(&As[cur][r][laneHi * 8]);
            af.u[1] = *(const uint4*)(&As[cur][r][16 + laneHi * 8]);
        }
        for (int sub = 0; sub < 8; ++sub) {
            int n = nHalf * 128 + sub * 16 + laneRow;
            AV bf;
            bf.u[0] = *(const uint4*)(&Bs[cur][n][laneHi * 8]);
            bf.u[1] = *(const uint4*)(&Bs[cur][n][16 + laneHi * 8]);
            acc[sub] = __builtin_amdgcn_wmma_f32_16x16x32_bf16(
                false, af.v, false, bf.v, (short)0, acc[sub], false, false);
        }
        __syncthreads();   // protect buffer `cur` for the step+2 issue
    }

    for (int sub = 0; sub < 8; ++sub) {
        int n = nHalf * 128 + sub * 16 + laneRow;
        float bv = bias[n];
        for (int v = 0; v < 8; ++v) {
            int m = v + laneHi * 8;
            float x = acc[sub][v] + bv;
            if (doRelu) x = fmaxf(x, 0.f);
            int gr = rowBase + mSub * 16 + m;
            Out[(size_t)gr * outRowScale * 256 + n] = (bf16_t)x;
        }
    }
}

// ---------------- GATv2 attention + softmax + residual + LayerNorm ----------------
__global__ __launch_bounds__(256) void gatv2_kernel(
    const bf16_t* __restrict__ XL, const bf16_t* __restrict__ XR,
    bf16_t* __restrict__ Xb, const float* __restrict__ att,
    const float* __restrict__ bias, const float* __restrict__ ln_g,
    const float* __restrict__ ln_b)
{
    int wave = threadIdx.x >> 5, lane = threadIdx.x & 31;
    int p = blockIdx.x * 8 + wave;
    size_t base = (size_t)p * NNODE * CDIM;
    int c0 = lane * 8;

    float xl[4][8], xr[4][8], xo[4][8], at[8];
    for (int c = 0; c < 8; ++c) at[c] = att[c0 + c];
    for (int i = 0; i < 4; ++i) {
        U8 a, b, o;
        a.u = *(const uint4*)(XL + base + i * CDIM + c0);
        b.u = *(const uint4*)(XR + base + i * CDIM + c0);
        o.u = *(const uint4*)(Xb + base + i * CDIM + c0);
        for (int c = 0; c < 8; ++c) {
            xl[i][c] = (float)a.h[c];
            xr[i][c] = (float)b.h[c];
            xo[i][c] = (float)o.h[c];
        }
    }
    float s[4][4];
    for (int i = 0; i < 4; ++i)
        for (int j = 0; j < 4; ++j) {
            float part = 0.f;
            for (int c = 0; c < 8; ++c) {
                float e = xr[i][c] + xl[j][c];
                e = (e > 0.f) ? e : 0.2f * e;
                part += at[c] * e;
            }
            part += __shfl_xor(part, 1, 32);
            part += __shfl_xor(part, 2, 32);
            part += __shfl_xor(part, 4, 32);
            s[i][j] = part;
        }
    float alpha[4][4];
    for (int i = 0; i < 4; ++i) {
        float mx = fmaxf(fmaxf(s[i][0], s[i][1]), fmaxf(s[i][2], s[i][3]));
        float den = 0.f;
        for (int j = 0; j < 4; ++j) { alpha[i][j] = __expf(s[i][j] - mx); den += alpha[i][j]; }
        float inv = 1.f / den;
        for (int j = 0; j < 4; ++j) alpha[i][j] *= inv;
    }
    for (int i = 0; i < 4; ++i) {
        float nw[8];
        float sum = 0.f, sq = 0.f;
        for (int c = 0; c < 8; ++c) {
            float o = alpha[i][0] * xl[0][c] + alpha[i][1] * xl[1][c]
                    + alpha[i][2] * xl[2][c] + alpha[i][3] * xl[3][c];
            nw[c] = o + bias[c0 + c] + xo[i][c];
            sum += nw[c]; sq += nw[c] * nw[c];
        }
        for (int off = 1; off < 32; off <<= 1) {
            sum += __shfl_xor(sum, off, 32);
            sq  += __shfl_xor(sq,  off, 32);
        }
        float mu = sum * (1.f / 256.f);
        float var = sq * (1.f / 256.f) - mu * mu;
        float inv = rsqrtf(var + 1e-5f);
        U8 o;
        for (int c = 0; c < 8; ++c)
            o.h[c] = (bf16_t)((nw[c] - mu) * inv * ln_g[c0 + c] + ln_b[c0 + c]);
        *(uint4*)(Xb + base + i * CDIM + c0) = o.u;
    }
}

// ---------------- zero-padded NHWC buffer from node-0 rows ----------------
__global__ __launch_bounds__(256) void pad_kernel(
    const bf16_t* __restrict__ Xb, bf16_t* __restrict__ Pad)
{
    size_t idx = (size_t)blockIdx.x * 256 + threadIdx.x;
    int c = (int)(idx & 255);
    size_t r = idx >> 8;
    int wp = (int)(r % PADW); r /= PADW;
    int hp = (int)(r % PADH);
    int b  = (int)(r / PADH);
    bf16_t v = (bf16_t)0.f;
    if (wp >= 1 && wp <= WW && hp >= 1 && hp <= HH) {
        int p = (b * HH + (hp - 1)) * WW + (wp - 1);
        v = Xb[(size_t)p * NNODE * CDIM + c];
    }
    Pad[idx] = v;
}

#if HAVE_TDM
__device__ __forceinline__ void conv_issue_stage(
    const bf16_t* Pad, const bf16_t* Ct, int b, int h, int w0, int s,
    unsigned ldsA, unsigned ldsB)
{
    int tap = s >> 3, k0 = (s & 7) * 32;
    int ky = tap / 3, kx = tap - ky * 3;
    size_t padRow = ((size_t)(b * PADH + h + ky) * PADW) + (w0 + kx);
    tdm_load_2d(Pad + padRow * 256 + k0, ldsA, 256, 0x40000000u, 32, 64, 256);
    tdm_load_2d(Ct + (size_t)tap * 65536 + k0, ldsB, 256, 256, 32, 256, 256);
}
#endif

// ---------------- conv3x3 implicit GEMM + BN + ReLU + residual -> NCHW f32 ----------------
// 72 pipeline stages (9 taps x 8 K-steps), double-buffered LDS + async TDM.
__global__ __launch_bounds__(256) void conv_bn_kernel(
    const bf16_t* __restrict__ Pad, const bf16_t* __restrict__ Ct,
    const bf16_t* __restrict__ Xb, const float* __restrict__ bn_g,
    const float* __restrict__ bn_b, const float* __restrict__ bn_m,
    const float* __restrict__ bn_v, float* __restrict__ out)
{
    __shared__ __align__(16) bf16_t As[2][64][32];
    __shared__ __align__(16) bf16_t Bs[2][256][32];
    int tid = threadIdx.x;
    int lane = tid & 31;
    int mSub = (tid >> 5) & 3, nHalf = tid >> 7;
    int laneRow = lane & 15, laneHi = lane >> 4;
    int pixBase = blockIdx.x * 64;
    int b  = pixBase >> 14;
    int h  = (pixBase >> 7) & 127;
    int w0 = pixBase & 127;

    v8f acc[8];
    for (int s = 0; s < 8; ++s) acc[s] = v8f{0.f,0.f,0.f,0.f,0.f,0.f,0.f,0.f};

#if HAVE_TDM
    if (tid == 0)
        conv_issue_stage(Pad, Ct, b, h, w0, 0,
                         (unsigned)(unsigned long long)&As[0][0][0],
                         (unsigned)(unsigned long long)&Bs[0][0][0]);
#endif
    for (int step = 0; step < 72; ++step) {
        int cur = step & 1;
#if HAVE_TDM
        if (tid == 0) {
            if (step + 1 < 72) {
                conv_issue_stage(Pad, Ct, b, h, w0, step + 1,
                                 (unsigned)(unsigned long long)&As[1 - cur][0][0],
                                 (unsigned)(unsigned long long)&Bs[1 - cur][0][0]);
                __builtin_amdgcn_s_wait_tensorcnt(2);
            } else {
                __builtin_amdgcn_s_wait_tensorcnt(0);
            }
        }
#else
        {
            int tap = step >> 3, k0 = (step & 7) * 32;
            int ky = tap / 3, kx = tap - ky * 3;
            size_t padRow = ((size_t)(b * PADH + h + ky) * PADW) + (w0 + kx);
            int r = tid >> 2, cpos = (tid & 3) * 8;
            *(uint4*)&As[cur][r][cpos] =
                *(const uint4*)(Pad + (padRow + r) * 256 + k0 + cpos);
            const bf16_t* bsrc = Ct + (size_t)tap * 65536 + (size_t)tid * 256 + k0;
            *(uint4*)&Bs[cur][tid][0]  = *(const uint4*)(bsrc);
            *(uint4*)&Bs[cur][tid][8]  = *(const uint4*)(bsrc + 8);
            *(uint4*)&Bs[cur][tid][16] = *(const uint4*)(bsrc + 16);
            *(uint4*)&Bs[cur][tid][24] = *(const uint4*)(bsrc + 24);
        }
#endif
        __syncthreads();

        AV af;
        {
            int r = mSub * 16 + laneRow;
            af.u[0] = *(const uint4*)(&As[cur][r][laneHi * 8]);
            af.u[1] = *(const uint4*)(&As[cur][r][16 + laneHi * 8]);
        }
        for (int sub = 0; sub < 8; ++sub) {
            int n = nHalf * 128 + sub * 16 + laneRow;
            AV bf;
            bf.u[0] = *(const uint4*)(&Bs[cur][n][laneHi * 8]);
            bf.u[1] = *(const uint4*)(&Bs[cur][n][16 + laneHi * 8]);
            acc[sub] = __builtin_amdgcn_wmma_f32_16x16x32_bf16(
                false, af.v, false, bf.v, (short)0, acc[sub], false, false);
        }
        __syncthreads();
    }
    for (int sub = 0; sub < 8; ++sub) {
        int o = nHalf * 128 + sub * 16 + laneRow;
        float scl = rsqrtf(bn_v[o] + 1e-5f) * bn_g[o];
        float sft = bn_b[o] - bn_m[o] * scl;
        for (int v = 0; v < 8; ++v) {
            int m = v + laneHi * 8;
            int pix = pixBase + mSub * 16 + m;
            float fused = (float)Xb[(size_t)pix * NNODE * CDIM + o];
            float y = acc[sub][v] * scl + sft;
            y = fmaxf(y, 0.f) + fused;
            int bb = pix >> 14, rem = pix & (HWSZ - 1);
            out[(size_t)bb * CDIM * HWSZ + (size_t)o * HWSZ + rem] = y;
        }
    }
}

extern "C" void kernel_launch(void* const* d_in, const int* in_sizes, int n_in,
                              void* d_out, int out_size, void* d_ws, size_t ws_size,
                              hipStream_t stream) {
    const float* modal0  = (const float*)d_in[0];
    const float* modal1  = (const float*)d_in[1];
    const float* modal2  = (const float*)d_in[2];
    const float* fn_w1   = (const float*)d_in[3];
    const float* fn_b1   = (const float*)d_in[4];
    const float* fn_w2   = (const float*)d_in[5];
    const float* fn_b2   = (const float*)d_in[6];
    const float* l0_wl   = (const float*)d_in[7];
    const float* l0_bl   = (const float*)d_in[8];
    const float* l0_wr   = (const float*)d_in[9];
    const float* l0_br   = (const float*)d_in[10];
    const float* l0_att  = (const float*)d_in[11];
    const float* l0_bias = (const float*)d_in[12];
    const float* ln0_g   = (const float*)d_in[13];
    const float* ln0_b   = (const float*)d_in[14];
    const float* l1_wl   = (const float*)d_in[15];
    const float* l1_bl   = (const float*)d_in[16];
    const float* l1_wr   = (const float*)d_in[17];
    const float* l1_br   = (const float*)d_in[18];
    const float* l1_att  = (const float*)d_in[19];
    const float* l1_bias = (const float*)d_in[20];
    const float* ln1_g   = (const float*)d_in[21];
    const float* ln1_b   = (const float*)d_in[22];
    const float* conv_w  = (const float*)d_in[23];
    const float* bn_g    = (const float*)d_in[24];
    const float* bn_b    = (const float*)d_in[25];
    const float* bn_m    = (const float*)d_in[26];
    const float* bn_v    = (const float*)d_in[27];

    char* ws = (char*)d_ws;
    bf16_t* Xb   = (bf16_t*)(ws + XB_OFF);
    bf16_t* XL   = (bf16_t*)(ws + XL_OFF);
    bf16_t* XR   = (bf16_t*)(ws + XR_OFF);
    bf16_t* Mean = (bf16_t*)(ws + MEAN_OFF);
    bf16_t* H1   = (bf16_t*)(ws + H1_OFF);
    bf16_t* Pad  = (bf16_t*)(ws + PAD_OFF);
    bf16_t* Wt   = (bf16_t*)(ws + WT_OFF);
    bf16_t* Ct   = Wt + (size_t)6 * 65536;
    float*  out  = (float*)d_out;

    wcvt_dense_kernel<<<dim3(256, 6), 256, 0, stream>>>(
        fn_w1, fn_w2, l0_wl, l0_wr, l1_wl, l1_wr, Wt);
    wcvt_conv_kernel<<<dim3(256, 9), 256, 0, stream>>>(conv_w, Ct);

    build_nodes_kernel<<<(NPIX * CDIM) / 256, 256, 0, stream>>>(
        modal0, modal1, modal2, Xb, Mean);

    gemm_bf16_kernel<<<NPIX / 64, 256, 0, stream>>>(Mean, Wt + 0 * 65536, fn_b1, H1, 1, 1);
    gemm_bf16_kernel<<<NPIX / 64, 256, 0, stream>>>(H1,   Wt + 1 * 65536, fn_b2, Xb, NNODE, 0);

    gemm_bf16_kernel<<<NROWS / 64, 256, 0, stream>>>(Xb, Wt + 2 * 65536, l0_bl, XL, 1, 0);
    gemm_bf16_kernel<<<NROWS / 64, 256, 0, stream>>>(Xb, Wt + 3 * 65536, l0_br, XR, 1, 0);
    gatv2_kernel<<<NPIX / 8, 256, 0, stream>>>(XL, XR, Xb, l0_att, l0_bias, ln0_g, ln0_b);

    gemm_bf16_kernel<<<NROWS / 64, 256, 0, stream>>>(Xb, Wt + 4 * 65536, l1_bl, XL, 1, 0);
    gemm_bf16_kernel<<<NROWS / 64, 256, 0, stream>>>(Xb, Wt + 5 * 65536, l1_br, XR, 1, 0);
    gatv2_kernel<<<NPIX / 8, 256, 0, stream>>>(XL, XR, Xb, l1_att, l1_bias, ln1_g, ln1_b);

    pad_kernel<<<(BATCH * PADH * PADW * CDIM) / 256, 256, 0, stream>>>(Xb, Pad);
    conv_bn_kernel<<<NPIX / 64, 256, 0, stream>>>(
        Pad, Ct, Xb, bn_g, bn_b, bn_m, bn_v, out);
}
